// KernelPointAggregation_29085518529282
// MI455X (gfx1250) — compile-verified, matrix operand order checked by reading, not run on hardware
//
#include <hip/hip_runtime.h>

#define N_PTS 8192
#define M_NEI 32
#define K_KP  8
#define D     64
#define EPSV  1e-7f

typedef __attribute__((ext_vector_type(16))) _Float16 v16h;
typedef __attribute__((ext_vector_type(8)))  _Float16 v8h;
typedef __attribute__((ext_vector_type(8)))  float    v8f;

union FragH { v16h v; v8h h[2]; };

// ---------------------------------------------------------------------------
// Pre-kernel: W (f32) -> Wh (f16) in workspace; compute x_kernel (8 x 64 f32).
// x_kernel[k<7] = expmap(e0, kernel_points[k+1]) (ptransp(e0,e0,.) is identity
// because kernel_points have zero time component); x_kernel[7] = e0.
// ---------------------------------------------------------------------------
__global__ __launch_bounds__(256) void kpa_prep(const float* __restrict__ W,
                                                const float* __restrict__ kp,
                                                _Float16* __restrict__ Wh,
                                                float* __restrict__ xk) {
  int t = blockIdx.x * blockDim.x + threadIdx.x;
  if (t < K_KP * D * D) Wh[t] = (_Float16)W[t];
  if (blockIdx.x == 0 && threadIdx.x < K_KP) {
    int k = threadIdx.x;
    if (k == K_KP - 1) {
      xk[k * D + 0] = 1.0f;
      for (int d2 = 1; d2 < D; ++d2) xk[k * D + d2] = 0.0f;
    } else {
      const float* row = kp + (k + 1) * D;
      float r2 = 0.f;
      for (int d2 = 1; d2 < D; ++d2) r2 += row[d2] * row[d2];
      float un = sqrtf(fmaxf(r2, 1e-14f));
      float sc = sinhf(un) / un;
      xk[k * D + 0] = coshf(un);
      for (int d2 = 1; d2 < D; ++d2) xk[k * D + d2] = sc * row[d2];
    }
  }
}

// ---------------------------------------------------------------------------
// Main kernel: one workgroup (256 threads = 8 wave32) per point n.
// ---------------------------------------------------------------------------
__global__ __launch_bounds__(256) void kpa_main(
    const float* __restrict__ x, const int* __restrict__ nei,
    const int* __restrict__ nmask, const float* __restrict__ bvec,
    const float* __restrict__ scale, const _Float16* __restrict__ Wh,
    const float* __restrict__ xk, float* __restrict__ out) {
  __shared__ float xrL[D];
  __shared__ int   nidxL[M_NEI];
  __shared__ float maskL[M_NEI];
  __shared__ float xneiL[M_NEI * 65];                 // padded: bank friendly
  __shared__ float x0fL[M_NEI * 65];                  // x0_nei in f32 (for dis)
  __shared__ __align__(16) _Float16 x0hL[M_NEI * 72]; // x0_nei in f16 (WMMA A)
  __shared__ float xkL[K_KP * D];
  __shared__ float disL[K_KP * M_NEI];
  __shared__ float hkL[M_NEI * 65];                   // one k-slice of h
  __shared__ float midL[M_NEI * 65];
  __shared__ float expE[K_KP];
  __shared__ float aveL[D];
  __shared__ float sgL[D];
  __shared__ float nrmL;

  const int n    = blockIdx.x;
  const int t    = threadIdx.x;
  const int lane = t & 31;
  const int wv   = t >> 5;
  const int m    = t >> 3;   // 8-lane group id: neighbor index
  const int sub  = t & 7;    // lane within group

  if (t < M_NEI) {
    nidxL[t] = nei[n * M_NEI + t];
    maskL[t] = (float)nmask[n * M_NEI + t];
  }
  if (t < K_KP) expE[t] = expf(scale[t]);
  if (t < D) xrL[t] = x[n * D + t];
  for (int i = t; i < K_KP * D; i += 256) xkL[i] = xk[i];
  for (int i = t; i < M_NEI * 65; i += 256) midL[i] = 0.f;
  __syncthreads();

  // Gather neighbor rows, coalesced (64 consecutive lanes per row).
  {
    int rowi = t >> 6;   // 0..3
    int d2   = t & 63;
    for (int r = 0; r < 8; ++r) {
      int mm = r * 4 + rowi;
      xneiL[mm * 65 + d2] = x[nidxL[mm] * D + d2];
    }
  }
  __syncthreads();

  // Phase 1: per-(m) logmap -> ptransp to origin -> expmap0 (groups of 8).
  {
    float aacc = 0.f;
    #pragma unroll
    for (int j = 0; j < 8; ++j) {
      int d2 = sub * 8 + j;
      float xv = xrL[d2], yv = xneiL[m * 65 + d2];
      aacc += (d2 == 0) ? xv * yv : -(xv * yv);   // alpha = -l_inner(x, y)
    }
    aacc += __shfl_xor(aacc, 1);
    aacc += __shfl_xor(aacc, 2);
    aacc += __shfl_xor(aacc, 4);
    float alpha = fmaxf(aacc, 1.0f + EPSV);
    float den   = sqrtf(fmaxf(alpha * alpha - 1.0f, 1e-14f));
    float dist  = acoshf(alpha);
    float coef  = dist / den;
    float x0r   = xrL[0];
    float y0    = xneiL[m * 65 + 0];
    float v0    = coef * (y0 - alpha * x0r);
    float factor = -v0 / (1.0f + x0r);            // ptransp coefficient
    float uloc[8];
    float usq = 0.f;
    #pragma unroll
    for (int j = 0; j < 8; ++j) {
      int d2 = sub * 8 + j;
      float u = coef * (xneiL[m * 65 + d2] - alpha * xrL[d2]) + factor * xrL[d2];
      uloc[j] = u;
      if (d2 != 0) usq += u * u;
    }
    usq += __shfl_xor(usq, 1);
    usq += __shfl_xor(usq, 2);
    usq += __shfl_xor(usq, 4);
    float un = sqrtf(fmaxf(usq, 1e-14f));
    float ch = coshf(un);
    float sc = sinhf(un) / un;
    #pragma unroll
    for (int j = 0; j < 8; ++j) {
      int d2 = sub * 8 + j;
      float xv = (d2 == 0) ? ch : sc * uloc[j];
      x0fL[m * 65 + d2] = xv;
      x0hL[m * 72 + d2] = (_Float16)xv;
    }
  }
  __syncthreads();

  // Phase 2: Lorentz distances to kernel points: wave wv -> k, lane -> m.
  {
    int k = wv, mm = lane;
    float acc = 0.f;
    #pragma unroll 4
    for (int d2 = 0; d2 < D; ++d2) {
      float xv = x0fL[mm * 65 + d2];
      float kv = xkL[k * D + d2];
      acc += (d2 == 0) ? xv * kv : -(xv * kv);
    }
    float alpha = fmaxf(acc, 1.0f + EPSV);
    disL[k * M_NEI + mm] = acoshf(alpha) * maskL[mm];
    // visibility guaranteed by the first barrier inside the k loop
  }

  // Phase 3-5: per-k GEMM slice via WMMA + transform + weighted accumulation.
  const int mt = wv >> 2;      // m-tile 0..1
  const int tc = wv & 3;       // o-tile 0..3
  const int cl = lane & 15;
  const int hi = lane >> 4;
  for (int k = 0; k < K_KP; ++k) {
    v8f acc = {};
    #pragma unroll
    for (int c = 0; c < 2; ++c) {
      FragH fa, fb;
      int arow = mt * 16 + cl;
      // A fragment (16x32 f16): lane<16 holds K {0..7,16..23}, lane>=16 {8..15,24..31}
      fa.h[0] = *(const v8h*)&x0hL[arow * 72 + c * 32 + hi * 8];
      fa.h[1] = *(const v8h*)&x0hL[arow * 72 + c * 32 + 16 + hi * 8];
      // B fragment: column ko = k*64 + tc*16 + cl, same K striping
      int ko = k * D + tc * 16 + cl;
      const _Float16* wp = Wh + ko * D + c * 32 + hi * 8;
      fb.h[0] = *(const v8h*)wp;
      fb.h[1] = *(const v8h*)(wp + 16);
      acc = __builtin_amdgcn_wmma_f32_16x16x32_f16(false, fa.v, false, fb.v,
                                                   (short)0, acc, false, false);
    }
    {
      int ko = k * D + tc * 16 + cl;
      float bv = bvec[ko];
      int o = tc * 16 + cl;
      #pragma unroll
      for (int v = 0; v < 8; ++v)
        hkL[(mt * 16 + hi * 8 + v) * 65 + o] = acc[v] + bv;
    }
    __syncthreads();
    {
      float vals[8];
      float nn = 0.f;
      #pragma unroll
      for (int j = 0; j < 8; ++j) {
        int o = sub * 8 + j;
        float hv = hkL[m * 65 + o];
        vals[j] = hv;
        if (o != 0) nn += hv * hv;
      }
      nn += __shfl_xor(nn, 1);
      nn += __shfl_xor(nn, 2);
      nn += __shfl_xor(nn, 4);
      float h0  = hkL[m * 65 + 0];
      float tme = 1.0f / (1.0f + expf(-h0)) * expE[k] + 1.0001f;
      float s   = (tme * tme - 1.0f) / fmaxf(nn, 1e-8f);
      float sq  = sqrtf(s);
      float dk  = disL[k * M_NEI + m];
      #pragma unroll
      for (int j = 0; j < 8; ++j) {
        int o = sub * 8 + j;
        float xt = (o == 0) ? tme : vals[j] * sq;
        midL[m * 65 + o] += dk * xt;   // exclusive per-thread slots
      }
    }
    __syncthreads();
  }

  // Per-(n,m) Lorentz normalization of mid.
  {
    float vals[8];
    float ns = 0.f;
    #pragma unroll
    for (int j = 0; j < 8; ++j) {
      int o = sub * 8 + j;
      float mv = midL[m * 65 + o];
      vals[j] = mv;
      ns += (o == 0) ? -(mv * mv) : mv * mv;
    }
    ns += __shfl_xor(ns, 1);
    ns += __shfl_xor(ns, 2);
    ns += __shfl_xor(ns, 4);
    float inv = 1.0f / sqrtf(fmaxf(fabsf(ns), 1e-8f));
    #pragma unroll
    for (int j = 0; j < 8; ++j) {
      int o = sub * 8 + j;
      midL[m * 65 + o] = vals[j] * inv;
    }
  }
  __syncthreads();

  // Mean over neighbors + final Lorentz normalization.
  if (t < D) {
    float a = 0.f;
    for (int mm = 0; mm < M_NEI; ++mm) a += midL[mm * 65 + t];
    a *= (1.0f / M_NEI);
    aveL[t] = a;
    sgL[t]  = (t == 0) ? -(a * a) : a * a;
  }
  __syncthreads();
  if (t == 0) {
    float ss = 0.f;
    for (int o = 0; o < D; ++o) ss += sgL[o];
    nrmL = sqrtf(fmaxf(fabsf(ss), 1e-8f));
  }
  __syncthreads();
  if (t < D) out[n * D + t] = aveL[t] / nrmL;
}

extern "C" void kernel_launch(void* const* d_in, const int* in_sizes, int n_in,
                              void* d_out, int out_size, void* d_ws, size_t ws_size,
                              hipStream_t stream) {
  const float* x     = (const float*)d_in[0];
  const int*   nei   = (const int*)d_in[1];
  const int*   nmask = (const int*)d_in[2];
  const float* W     = (const float*)d_in[3];
  const float* b     = (const float*)d_in[4];
  const float* scale = (const float*)d_in[5];
  const float* kp    = (const float*)d_in[6];

  _Float16* Wh = (_Float16*)d_ws;
  float*    xk = (float*)((char*)d_ws + (size_t)(K_KP * D * D) * sizeof(_Float16));

  kpa_prep<<<(K_KP * D * D + 255) / 256, 256, 0, stream>>>(W, kp, Wh, xk);
  kpa_main<<<N_PTS, 256, 0, stream>>>(x, nei, nmask, b, scale, Wh, xk,
                                      (float*)d_out);
}
